// SRUCell_90632399880536
// MI455X (gfx1250) — compile-verified, matrix operand order checked by reading, not run on hardware
//
#include <hip/hip_runtime.h>
#include <math.h>

// ---------------- problem constants (from reference) ----------------
#define LSEQ 1024
#define BATCH 8
#define DIM 2048
#define MTOT (LSEQ * BATCH)      // 8192 GEMM rows
#define NTOT (3 * DIM)           // 6144 GEMM cols
#define KTOT DIM                 // 2048 reduction

#define SCALE_X 1.7320508075688772f   // sqrt(1 + 2*exp(0))

// ---------------- GEMM tiling ----------------
#define BM 128
#define BN 128
#define BK 32
#define NK (KTOT / BK)           // 64 K-steps
#define ROWU 36                  // LDS row stride in ushorts (72B: 8B-aligned, conflict-benign)

typedef __attribute__((ext_vector_type(16))) __bf16 v16bf;
typedef __attribute__((ext_vector_type(8)))  float  v8f;

union Frag16 {
    v16bf        v;
    unsigned int w[8];
};

__device__ __forceinline__ unsigned short f2bf_rne(float f) {
    unsigned int u = __float_as_uint(f);
    u += 0x7FFFu + ((u >> 16) & 1u);       // round-to-nearest-even
    return (unsigned short)(u >> 16);
}

// ---------------- async global->LDS (guarded; fallback = sync b64 copy) ----
#if defined(__has_builtin)
# if __has_builtin(__builtin_amdgcn_global_load_async_to_lds_b64)
#  define HAVE_ASYNC_LDS 1
# endif
#endif
#ifndef HAVE_ASYNC_LDS
# define HAVE_ASYNC_LDS 0
#endif

typedef int v2i_vs __attribute__((vector_size(8)));
typedef __attribute__((address_space(1))) v2i_vs* g_v2i_p;
typedef __attribute__((address_space(3))) v2i_vs* l_v2i_p;

__device__ __forceinline__ void async_copy_b64(const void* g, void* l) {
#if HAVE_ASYNC_LDS
    __builtin_amdgcn_global_load_async_to_lds_b64(
        (g_v2i_p)(unsigned long long)g,
        (l_v2i_p)(unsigned int)(unsigned long long)l,
        0, 0);
#else
    *(uint2*)l = *(const uint2*)g;
#endif
}

__device__ __forceinline__ void wait_async_all() {
#if HAVE_ASYNC_LDS
# if __has_builtin(__builtin_amdgcn_s_wait_asynccnt)
    __builtin_amdgcn_s_wait_asynccnt(0);
# else
    asm volatile("s_wait_asynccnt 0x0" ::: "memory");
# endif
#endif
}

// ---------------- one-time precision conversion kernels ----------------
// x (f32, MTOT*KTOT) -> packed bf16 pairs
__global__ __launch_bounds__(256)
void cvt_bf16_pairs(const float* __restrict__ src, unsigned int* __restrict__ dst,
                    int nPairs) {
    int i = blockIdx.x * blockDim.x + threadIdx.x;
    if (i < nPairs) {
        float2 v = ((const float2*)src)[i];
        dst[i] = (unsigned int)f2bf_rne(v.x) | ((unsigned int)f2bf_rne(v.y) << 16);
    }
}

// W (f32, KTOT x NTOT) -> WT (bf16, NTOT x KTOT)  (transpose + convert, LDS tiled)
__global__ __launch_bounds__(256)
void cvt_w_transpose(const float* __restrict__ W, unsigned short* __restrict__ WT) {
    __shared__ unsigned short t[64][65];
    const int tid = threadIdx.x;
    const int n0 = blockIdx.x * 64;       // over NTOT (96 blocks)
    const int k0 = blockIdx.y * 64;       // over KTOT (32 blocks)

    #pragma unroll
    for (int i = 0; i < 16; ++i) {        // 64*64 / 256
        int lin = i * 256 + tid;
        int kk = lin >> 6, nn = lin & 63;
        t[nn][kk] = f2bf_rne(W[(size_t)(k0 + kk) * NTOT + (n0 + nn)]);
    }
    __syncthreads();
    #pragma unroll
    for (int i = 0; i < 8; ++i) {         // 2048 uint-pairs / 256
        int lin = i * 256 + tid;
        int nn = lin >> 5, kk2 = (lin & 31) << 1;
        unsigned int v = (unsigned int)t[nn][kk2] | ((unsigned int)t[nn][kk2 + 1] << 16);
        *(unsigned int*)&WT[(size_t)(n0 + nn) * KTOT + k0 + kk2] = v;
    }
}

// ---------------- WMMA GEMM: U = XB(bf16) * WT(bf16)^T, f32 accumulate ----
// 8 waves: 4(M) x 2(N); wave tile 32x64 = 2x4 WMMA 16x16 tiles.
__global__ __launch_bounds__(256)
void sru_gemm_wmma(const unsigned short* __restrict__ XB,   // (MTOT, KTOT) bf16
                   const unsigned short* __restrict__ WT,   // (NTOT, KTOT) bf16
                   float* __restrict__ U)                   // 3 planes (MTOT, DIM)
{
    __shared__ unsigned short As[2][BM][ROWU];
    __shared__ unsigned short Bs[2][BN][ROWU];

    const int tid    = threadIdx.x;
    const int lane   = tid & 31;
    const int waveId = tid >> 5;
    const int waveM  = waveId & 3;                  // 0..3 -> 32 rows each
    const int waveN  = waveId >> 2;                 // 0..1 -> 64 cols each
    const int half   = lane >> 4;
    const int lm     = lane & 15;

    const int blockM = blockIdx.x * BM;
    const int blockN = blockIdx.y * BN;

    v8f acc[2][4] = {};

    // per-thread fill assignment: 4 chunks of b64 for each of A and B per K-step
    int rowC[4], kbC[4];
    #pragma unroll
    for (int c = 0; c < 4; ++c) {
        int lin = c * 256 + tid;                    // 1024 b64 chunks / tile
        rowC[c] = lin >> 3;                         // 0..127
        kbC[c]  = (lin & 7) << 3;                   // byte offset in 64B row
    }

    // issue fills for a K-step into buffer p
    auto fill = [&](int p, int kt) {
        #pragma unroll
        for (int c = 0; c < 4; ++c) {
            const char* ga = (const char*)(XB + (size_t)(blockM + rowC[c]) * KTOT + kt * BK) + kbC[c];
            async_copy_b64(ga, (char*)&As[p][rowC[c]][0] + kbC[c]);
            const char* gb = (const char*)(WT + (size_t)(blockN + rowC[c]) * KTOT + kt * BK) + kbC[c];
            async_copy_b64(gb, (char*)&Bs[p][rowC[c]][0] + kbC[c]);
        }
    };

    int p = 0;
    fill(p, 0);
    wait_async_all();
    __syncthreads();

    for (int kt = 0; kt < NK; ++kt) {
        if (kt + 1 < NK) fill(p ^ 1, kt + 1);       // prefetch next tile (other buffer)

        // ---- build fragments per ISA 16-bit A 16x32 layout ----
        Frag16 a[2], b[4];
        #pragma unroll
        for (int v = 0; v < 8; ++v) {
            const int kb = ((v >> 2) << 4) + (half << 3) + ((v & 3) << 1);
            a[0].w[v] = *(const unsigned int*)&As[p][waveM * 32      + lm][kb];
            a[1].w[v] = *(const unsigned int*)&As[p][waveM * 32 + 16 + lm][kb];
            #pragma unroll
            for (int tn = 0; tn < 4; ++tn)
                b[tn].w[v] = *(const unsigned int*)&Bs[p][waveN * 64 + tn * 16 + lm][kb];
        }

        #pragma unroll
        for (int tm = 0; tm < 2; ++tm)
            #pragma unroll
            for (int tn = 0; tn < 4; ++tn)
                acc[tm][tn] = __builtin_amdgcn_wmma_f32_16x16x32_bf16(
                    false, a[tm].v, false, b[tn].v,
                    (short)0, acc[tm][tn], false, false);

        wait_async_all();
        __syncthreads();
        p ^= 1;
    }

    // ---- store: de-interleave column n = 3*d + j into plane j ----
    const size_t planeStride = (size_t)MTOT * DIM;
    #pragma unroll
    for (int tm = 0; tm < 2; ++tm) {
        #pragma unroll
        for (int tn = 0; tn < 4; ++tn) {
            #pragma unroll
            for (int r = 0; r < 8; ++r) {
                int gm = blockM + waveM * 32 + tm * 16 + r + (half << 3);
                int gn = blockN + waveN * 64 + tn * 16 + lm;
                int d = gn / 3;
                int j = gn - 3 * d;
                U[(size_t)j * planeStride + (size_t)gm * DIM + d] = acc[tm][tn][r];
            }
        }
    }
}

// ---------------- sequential SRU recurrence ----------------
__global__ __launch_bounds__(256)
void sru_recurrence(const float* __restrict__ U,      // 3 planes (MTOT, DIM)
                    const float* __restrict__ X,      // (MTOT, DIM) f32
                    const float* __restrict__ C0,     // (BATCH, DIM)
                    const float* __restrict__ Wc,     // (2*DIM)
                    const float* __restrict__ Bias,   // (2*DIM)
                    float* __restrict__ Out)          // h | c_last
{
    const int idx = blockIdx.x * blockDim.x + threadIdx.x;
    if (idx >= BATCH * DIM) return;
    const int b = idx / DIM;
    const int d = idx - b * DIM;

    const size_t planeStride = (size_t)MTOT * DIM;
    const float vf = Wc[d];
    const float vr = Wc[DIM + d];
    const float bf = Bias[d];
    const float br = Bias[DIM + d];

    float c = C0[idx];
    const float* u0p = U;
    const float* u1p = U + planeStride;
    const float* u2p = U + 2 * planeStride;

    for (int l = 0; l < LSEQ; ++l) {
        const size_t m = (size_t)(l * BATCH + b) * DIM + d;
        const float u0 = u0p[m];
        const float u1 = u1p[m];
        const float u2 = u2p[m];
        const float xv = X[m];

        const float f = 1.0f / (1.0f + __expf(-(u1 + vf * c + bf)));
        c = (c - u0) * f + u0;
        const float r = 1.0f / (1.0f + __expf(-(u2 + vr * c + br)));
        Out[m] = r * c + (1.0f - r) * xv * SCALE_X;
    }
    Out[planeStride + idx] = c;
}

extern "C" void kernel_launch(void* const* d_in, const int* in_sizes, int n_in,
                              void* d_out, int out_size, void* d_ws, size_t ws_size,
                              hipStream_t stream) {
    (void)in_sizes; (void)n_in; (void)out_size; (void)ws_size;
    const float* x   = (const float*)d_in[0];   // (L,B,D)
    const float* c0  = (const float*)d_in[1];   // (B,D)
    const float* wgt = (const float*)d_in[2];   // (D,3D)
    const float* wc  = (const float*)d_in[3];   // (2D)
    const float* bia = (const float*)d_in[4];   // (2D)

    // workspace layout: U (192MB f32) | XB (32MB bf16) | WT (24MB bf16)
    float* U = (float*)d_ws;
    unsigned short* XB = (unsigned short*)((char*)d_ws + (size_t)3 * MTOT * DIM * sizeof(float));
    unsigned short* WT = XB + (size_t)MTOT * KTOT;

    // 1) one-time bf16 conversion of x
    const int nPairs = (MTOT * KTOT) / 2;           // 8,388,608
    cvt_bf16_pairs<<<(nPairs + 255) / 256, 256, 0, stream>>>(x, (unsigned int*)XB, nPairs);

    // 2) one-time convert + transpose of W -> (NTOT, KTOT)
    dim3 wGrid(NTOT / 64, KTOT / 64);               // 96 x 32
    cvt_w_transpose<<<wGrid, 256, 0, stream>>>(wgt, WT);

    // 3) WMMA GEMM
    dim3 gemmGrid(MTOT / BM, NTOT / BN);            // 64 x 48
    sru_gemm_wmma<<<gemmGrid, 256, 0, stream>>>(XB, WT, U);

    // 4) sequential recurrence
    const int nElem = BATCH * DIM;                  // 16384
    sru_recurrence<<<(nElem + 255) / 256, 256, 0, stream>>>(
        U, x, c0, wc, bia, (float*)d_out);
}
